// CrossGraphAttentionModule_26199300506297
// MI455X (gfx1250) — matrix operand order repaired and checked by reading, not compiled, not run on hardware
//
#include <hip/hip_runtime.h>
#include <math.h>

typedef __bf16 bf16;
typedef __attribute__((ext_vector_type(16))) __bf16 v16bf;
typedef __attribute__((ext_vector_type(8)))  __bf16 v8bf;
typedef __attribute__((ext_vector_type(8)))  float  v8f;

#define AN 16384
#define RN 8192
#define DD 128
#define NB 32
#define WPB 8             // waves per block (wave32)
#define CHUNK 64          // residues per pipeline stage
#define KPITCH 272        // LDS bytes per K row: 256 data + 16 pad (bank-conflict-free)
#define NEGBIG (-3.0e38f)

static __device__ __forceinline__ v8f wmma_bf16(v16bf a, v16bf b, v8f c) {
  return __builtin_amdgcn_wmma_f32_16x16x32_bf16(false, a, false, b, (short)0, c,
                                                 false, false);
}

// A-operand fragment of a 16x32 bf16 tile. Lane group g=0 (lanes 0-15) holds
// K = 0..15; g=1 holds K = 16..31 (contiguous). `row` = this lane's M-row
// at the K-chunk start.
static __device__ __forceinline__ v16bf load_a_frag(const bf16* row, int g) {
  v8bf lo = *(const v8bf*)(row + g * 16);
  v8bf hi = *(const v8bf*)(row + g * 16 + 8);
  v16bf f;
#pragma unroll
  for (int i = 0; i < 8; ++i) { f[i] = lo[i]; f[i + 8] = hi[i]; }
  return f;
}

// B-operand fragment of a 32x16 bf16 tile: lane (l&15) = column N, group g
// holds K = g*16..g*16+15 contiguously. 32B-aligned source.
static __device__ __forceinline__ v16bf load_b_frag(const bf16* col_k0, int g) {
  return *(const v16bf*)(col_k0 + g * 16);
}

// Same, but via two 16B loads (for 16B-aligned LDS rows with 272B pitch).
static __device__ __forceinline__ v16bf load_b_frag16(const bf16* col_k0, int g) {
  v8bf lo = *(const v8bf*)(col_k0 + g * 16);
  v8bf hi = *(const v8bf*)(col_k0 + g * 16 + 8);
  v16bf f;
#pragma unroll
  for (int i = 0; i < 8; ++i) { f[i] = lo[i]; f[i + 8] = hi[i]; }
  return f;
}

// ---------------------------------------------------------------------------
// Tensor Data Mover: stage a 2D bf16 tile (tile_w x tile_h) into LDS with
// row padding. D# per CDNA5 ISA ch.8 (count=1, type=2 "image", data_size=2B).
// This toolchain's builtin takes 6 args (g0, g1, g2, g3, extra, cpol).
// ---------------------------------------------------------------------------
#if defined(__has_builtin)
#if __has_builtin(__builtin_amdgcn_tensor_load_to_lds)
#define HAVE_TDM 1
#endif
#endif

#ifdef HAVE_TDM
typedef __attribute__((ext_vector_type(4))) unsigned int v4u;
typedef __attribute__((ext_vector_type(8))) int v8i;
typedef __attribute__((ext_vector_type(4))) int v4i;

static __device__ __forceinline__ void tdm_load_tile_bf16(
    void* lds_dst, const bf16* gsrc,
    unsigned tile_w, unsigned tile_h,
    unsigned tensor_w, unsigned tensor_h,
    unsigned long long row_stride_elems,
    unsigned pad_interval_code, unsigned pad_amount_code) {
  unsigned lds = (unsigned)(size_t)lds_dst;            // low 32b = LDS offset
  unsigned long long ga = (unsigned long long)(size_t)gsrc;
  v4u g0;
  g0.x = 1u;                                           // count=1, user D#
  g0.y = lds;                                          // lds_addr [63:32]
  g0.z = (unsigned)ga;                                 // global_addr [95:64]
  g0.w = (unsigned)((ga >> 32) & 0x1FFFFFFull) |       // global_addr [120:96]
         (2u << 30);                                   // type=2 [127:126]
  v8i g1;
  g1[0] = (int)((1u << 16) |                           // data_size = 2 bytes
                (1u << 20) |                           // pad_enable
                (pad_interval_code << 22) |
                (pad_amount_code << 25));
  g1[1] = (int)((tensor_w & 0xFFFFu) << 16);           // tensor_dim0 [79:48]
  g1[2] = (int)((tensor_w >> 16) | ((tensor_h & 0xFFFFu) << 16)); // dim1 [111:80]
  g1[3] = (int)((tensor_h >> 16) | ((tile_w & 0xFFFFu) << 16));   // tile0 [127:112]
  g1[4] = (int)(tile_h & 0xFFFFu);                     // tile_dim1 [143:128]
  g1[5] = (int)(row_stride_elems & 0xFFFFFFFFull);     // dim0_stride [207:160]
  g1[6] = (int)((row_stride_elems >> 32) & 0xFFFFull);
  g1[7] = 0;
  v4i z4 = (v4i)0;
  v8i z8 = (v8i)0;
  __builtin_amdgcn_tensor_load_to_lds(g0, g1, z4, z4, z8, 0);
}
#endif

// ---------------------------------------------------------------------------
// Per-batch residue segment bounds (residue_batch is sorted).
// ---------------------------------------------------------------------------
__global__ void seg_init_kernel(int* seg_lo, int* seg_hi) {
  int b = threadIdx.x;
  if (b < NB) { seg_lo[b] = RN; seg_hi[b] = 0; }
}

__global__ void seg_build_kernel(const int* __restrict__ residue_batch,
                                 int* seg_lo, int* seg_hi) {
  int i = blockIdx.x * blockDim.x + threadIdx.x;
  if (i < RN) {
    int b = residue_batch[i];
    atomicMin(&seg_lo[b], i);
    atomicMax(&seg_hi[b], i + 1);
  }
}

// ---------------------------------------------------------------------------
// Projection: Y = (X @ W^T) * out_scale, cast to bf16.
// ---------------------------------------------------------------------------
template <bool TRANS_OUT>
__global__ __launch_bounds__(256) void proj_kernel(const float* __restrict__ X,
                                                   const float* __restrict__ W,
                                                   bf16* __restrict__ Y,
                                                   int M, float out_scale) {
  int wave = (int)((blockIdx.x * blockDim.x + threadIdx.x) >> 5);
  int lane = threadIdx.x & 31;
  int g = lane >> 4;
  int lr = lane & 15;
  int m0 = wave * 16;
  if (m0 >= M) return;

  v16bf a[4];
  const float* xrow = X + (size_t)(m0 + lr) * DD;
#pragma unroll
  for (int kc = 0; kc < 4; ++kc) {
    const float* p = xrow + kc * 32;
    v16bf f;
#pragma unroll
    for (int i = 0; i < 8; ++i) {
      f[i]     = (bf16)p[g * 16 + i];
      f[i + 8] = (bf16)p[g * 16 + 8 + i];
    }
    a[kc] = f;
  }

  for (int n0 = 0; n0 < DD; n0 += 16) {
    const float* wrow = W + (size_t)(n0 + lr) * DD;
    v8f c = {};
#pragma unroll
    for (int kc = 0; kc < 4; ++kc) {
      v16bf b;
#pragma unroll
      for (int i = 0; i < 16; ++i) b[i] = (bf16)wrow[kc * 32 + g * 16 + i];
      c = wmma_bf16(a[kc], b, c);
    }
#pragma unroll
    for (int v = 0; v < 8; ++v) {
      int m = m0 + v + 8 * g;
      int n = n0 + lr;
      bf16 val = (bf16)(c[v] * out_scale);
      if (TRANS_OUT) Y[(size_t)n * M + m] = val;
      else           Y[(size_t)m * DD + n] = val;
    }
  }
}

// ---------------------------------------------------------------------------
// Flash attention. One block = 128 atoms (8 waves x 16). The block walks a
// uniform residue range; K chunks are TDM-staged into double-buffered LDS.
// ---------------------------------------------------------------------------
__global__ __launch_bounds__(256) void attn_kernel(
    const float* __restrict__ atom_h,
    const int* __restrict__ atom_batch,
    const int* __restrict__ residue_batch,
    const bf16* __restrict__ Qb,    // (A, 128) bf16, pre-scaled by log2e/sqrt(DH)
    const bf16* __restrict__ Kb,    // (R, 128) bf16
    const bf16* __restrict__ Vt,    // (128, R) bf16
    const int* __restrict__ seg_lo,
    const int* __restrict__ seg_hi,
    float* __restrict__ out) {
  __shared__ __align__(16) char Kl[2][CHUNK * KPITCH];   // 2 x 17 KB K stage
  __shared__ __align__(16) bf16 Ps[WPB][16 * CHUNK];     // per-wave P reshape

  int wid  = (int)(threadIdx.x >> 5);
  int lane = (int)(threadIdx.x & 31);
  int g  = lane >> 4;
  int lr = lane & 15;
  int a_blk = (int)blockIdx.x * (16 * WPB);
  int a0 = a_blk + wid * 16;

  // Resident Q fragments.
  v16bf q[4];
  const bf16* qrow = Qb + (size_t)(a0 + lr) * DD;
#pragma unroll
  for (int kc = 0; kc < 4; ++kc) q[kc] = load_a_frag(qrow + kc * 32, g);

  // Batch id of each C-layout row this lane owns (rows m = v + 8*g).
  int ab[8];
#pragma unroll
  for (int v = 0; v < 8; ++v) ab[v] = atom_batch[a0 + v + 8 * g];

  // Block-uniform residue range (atom_batch sorted).
  int b_lo = atom_batch[a_blk];
  int b_hi = atom_batch[a_blk + 16 * WPB - 1];
  int rlo = RN, rhi = 0;
  for (int b = b_lo; b <= b_hi; ++b) {
    rlo = min(rlo, seg_lo[b]);
    rhi = max(rhi, seg_hi[b]);
  }
  int rc0 = rlo & ~(CHUNK - 1);
  int nch = (rhi > rc0) ? ((rhi - rc0 + CHUNK - 1) / CHUNK) : 0;

  v8f ctx[8];
#pragma unroll
  for (int t = 0; t < 8; ++t) ctx[t] = (v8f){};
  float mrow[8], lrow[8];
#pragma unroll
  for (int v = 0; v < 8; ++v) { mrow[v] = NEGBIG; lrow[v] = 0.0f; }

#ifdef HAVE_TDM
  if (nch > 0 && wid == 0)
    tdm_load_tile_bf16(&Kl[0][0], Kb + (size_t)rc0 * DD,
                       DD, CHUNK, DD, RN, DD, /*ival 64dw*/5, /*pad 4dw*/3);
#endif

  for (int ic = 0; ic < nch; ++ic) {
    int rc = rc0 + ic * CHUNK;

#ifdef HAVE_TDM
    if (wid == 0) __builtin_amdgcn_s_wait_tensorcnt(0);
    __syncthreads();                       // publish chunk ic to all waves
    if (wid == 0 && ic + 1 < nch)
      tdm_load_tile_bf16(&Kl[(ic + 1) & 1][0], Kb + (size_t)(rc + CHUNK) * DD,
                         DD, CHUNK, DD, RN, DD, 5, 3);
#else
    __syncthreads();
    {   // cooperative copy fallback: 64 rows x 16 uint4
      const uint4* src = (const uint4*)(Kb + (size_t)rc * DD);
      char* dst = &Kl[ic & 1][0];
      for (int idx = (int)threadIdx.x; idx < CHUNK * 16; idx += 256) {
        int row = idx >> 4, col = idx & 15;
        *(uint4*)(dst + (size_t)row * KPITCH + col * 16) = src[idx];
      }
    }
    __syncthreads();
#endif
    const char* klbuf = &Kl[ic & 1][0];

    if (ic + 1 < nch)
      __builtin_prefetch(Vt + (size_t)(lane * 4) * RN + rc + CHUNK, 0, 0);

    // ---- scores: four 16x16 tiles from LDS-staged K ----------------------
    v8f s[4];
#pragma unroll
    for (int t = 0; t < 4; ++t) {
      const char* krow = klbuf + (size_t)(t * 16 + lr) * KPITCH;
      v8f c = {};
#pragma unroll
      for (int kc = 0; kc < 4; ++kc)
        c = wmma_bf16(q[kc], load_b_frag16((const bf16*)(krow + kc * 64), g), c);
      s[t] = c;
    }

    // ---- same-batch mask -------------------------------------------------
    int rb[4];
#pragma unroll
    for (int t = 0; t < 4; ++t) rb[t] = residue_batch[rc + t * 16 + lr];
#pragma unroll
    for (int t = 0; t < 4; ++t)
#pragma unroll
      for (int v = 0; v < 8; ++v)
        if (ab[v] != rb[t]) s[t][v] = NEGBIG;

    // ---- online softmax (row = one C VGPR across 16 lanes of a half) -----
#pragma unroll
    for (int v = 0; v < 8; ++v) {
      float mv = fmaxf(fmaxf(s[0][v], s[1][v]), fmaxf(s[2][v], s[3][v]));
#pragma unroll
      for (int off = 8; off; off >>= 1) mv = fmaxf(mv, __shfl_xor(mv, off, 16));
      float mnew = fmaxf(mrow[v], mv);
      float guard = (mnew > -1.0e37f) ? 1.0f : 0.0f;
      float corr = exp2f(mrow[v] - mnew);
      mrow[v] = mnew;
      float p0 = exp2f(s[0][v] - mnew) * guard;
      float p1 = exp2f(s[1][v] - mnew) * guard;
      float p2 = exp2f(s[2][v] - mnew) * guard;
      float p3 = exp2f(s[3][v] - mnew) * guard;
      s[0][v] = p0; s[1][v] = p1; s[2][v] = p2; s[3][v] = p3;
      float rs = (p0 + p1) + (p2 + p3);
#pragma unroll
      for (int off = 8; off; off >>= 1) rs += __shfl_xor(rs, off, 16);
      lrow[v] = lrow[v] * corr + rs;
#pragma unroll
      for (int t = 0; t < 8; ++t) ctx[t][v] *= corr;
    }

    // ---- reshape P (C-layout -> A-fragments) via per-wave LDS tile -------
    bf16* pl = &Ps[wid][0];
#pragma unroll
    for (int v = 0; v < 8; ++v) {
      int m = v + 8 * g;
#pragma unroll
      for (int t = 0; t < 4; ++t) pl[m * CHUNK + t * 16 + lr] = (bf16)s[t][v];
    }
    asm volatile("s_wait_dscnt 0x0" ::: "memory");
    v16bf pf0 = load_a_frag(pl + lr * CHUNK, g);
    v16bf pf1 = load_a_frag(pl + lr * CHUNK + 32, g);

    // ---- ctx += P @ V (B operand contiguous from transposed V) ----------
#pragma unroll
    for (int t = 0; t < 8; ++t) {
      const bf16* vcol = Vt + (size_t)(t * 16 + lr) * RN + rc;
      v8f c = ctx[t];
      c = wmma_bf16(pf0, load_b_frag(vcol, g), c);
      c = wmma_bf16(pf1, load_b_frag(vcol + 32, g), c);
      ctx[t] = c;
    }
  }

  // ---- epilogue: out = atom_h + ctx / l (rows with no residues: +0) ------
#pragma unroll
  for (int v = 0; v < 8; ++v) {
    float rl = (lrow[v] > 0.0f) ? (1.0f / lrow[v]) : 0.0f;
    int m = a0 + v + 8 * g;
#pragma unroll
    for (int t = 0; t < 8; ++t) {
      int idx = m * DD + t * 16 + lr;
      out[idx] = atom_h[idx] + ctx[t][v] * rl;
    }
  }
}

// ---------------------------------------------------------------------------
// Host launcher
// ---------------------------------------------------------------------------
extern "C" void kernel_launch(void* const* d_in, const int* in_sizes, int n_in,
                              void* d_out, int out_size, void* d_ws, size_t ws_size,
                              hipStream_t stream) {
  (void)in_sizes; (void)n_in; (void)out_size; (void)ws_size;

  const float* atom_h        = (const float*)d_in[0];
  const float* residue_h     = (const float*)d_in[1];
  const int*   atom_batch    = (const int*)d_in[2];
  const int*   residue_batch = (const int*)d_in[3];
  const float* W_q           = (const float*)d_in[4];
  const float* W_k           = (const float*)d_in[5];
  const float* W_v           = (const float*)d_in[6];
  float*       out           = (float*)d_out;

  char* ws = (char*)d_ws;
  size_t off = 0;
  bf16* Qb = (bf16*)(ws + off); off += (size_t)AN * DD * sizeof(bf16);  // 4 MiB
  bf16* Kb = (bf16*)(ws + off); off += (size_t)RN * DD * sizeof(bf16);  // 2 MiB
  bf16* Vt = (bf16*)(ws + off); off += (size_t)DD * RN * sizeof(bf16);  // 2 MiB
  int* seg_lo = (int*)(ws + off); off += NB * sizeof(int);
  int* seg_hi = (int*)(ws + off);

  seg_init_kernel<<<1, 32, 0, stream>>>(seg_lo, seg_hi);
  seg_build_kernel<<<RN / 256, 256, 0, stream>>>(residue_batch, seg_lo, seg_hi);

  const float qscale = 1.4426950408889634f / sqrtf((float)DD);
  proj_kernel<false><<<(AN / 16) / WPB, 256, 0, stream>>>(atom_h, W_q, Qb, AN, qscale);
  proj_kernel<false><<<(RN / 16) / WPB, 256, 0, stream>>>(residue_h, W_k, Kb, RN, 1.0f);
  proj_kernel<true ><<<(RN / 16) / WPB, 256, 0, stream>>>(residue_h, W_v, Vt, RN, 1.0f);

  attn_kernel<<<(AN / 16) / WPB, 256, 0, stream>>>(
      atom_h, atom_batch, residue_batch, Qb, Kb, Vt, seg_lo, seg_hi, out);
}
